// SimCLRLoss_2190433321272
// MI455X (gfx1250) — matrix operand order repaired and checked by reading, not compiled, not run on hardware
//
#include <hip/hip_runtime.h>
#include <hip/hip_bf16.h>
#include <math.h>
#include <stdint.h>

typedef __attribute__((ext_vector_type(16))) _Float16 v16h;
typedef __attribute__((ext_vector_type(8)))  _Float16 v8h;
typedef __attribute__((ext_vector_type(8)))  float    v8f;

#define N 8192
#define D 256
#define TEMP_INV 2.0f                 // 1 / TEMPERATURE
#define LOG2E 1.4426950408889634f
#define EPSN 1e-8f
#define ITILE 128                      // i-rows per block (8 waves x 16)
#define JCHUNK 1024                    // j-columns per block
#define JSTAGE 64                      // j-rows staged into LDS per step
#define LDSTRIDE 264                   // halves per row in LDS (256 + 8 pad -> bank rotate)

// ------------- Kernel 1: row-normalize, f16 hi/lo split, diag, zero rowsum -------------
__global__ __launch_bounds__(32) void k_normalize(const float* __restrict__ x,
                                                  _Float16* __restrict__ xhi,
                                                  _Float16* __restrict__ xlo,
                                                  float* __restrict__ diag,
                                                  float* __restrict__ rowsum)
{
    const int row = blockIdx.x;
    const int t = threadIdx.x;                 // 0..31, one wave per row
    const float* xr = x + (size_t)row * D;

    float v[8];
    float ss = 0.f;
    #pragma unroll
    for (int e = 0; e < 8; ++e) {
        v[e] = xr[t * 8 + e];
        ss = fmaf(v[e], v[e], ss);
    }
    #pragma unroll
    for (int m = 16; m >= 1; m >>= 1)
        ss += __shfl_xor(ss, m, 32);

    const float nrm = sqrtf(ss);
    const float inv = 1.0f / fmaxf(nrm, EPSN);

    _Float16* hr = xhi + (size_t)row * D + t * 8;
    _Float16* lr = xlo + (size_t)row * D + t * 8;
    #pragma unroll
    for (int e = 0; e < 8; ++e) {
        float xn = v[e] * inv;
        _Float16 h = (_Float16)xn;
        hr[e] = h;
        lr[e] = (_Float16)(xn - (float)h);
    }
    if (t == 0) {
        diag[row]   = ss * inv * inv;          // exact xn . xn (handles eps clamp)
        rowsum[row] = 0.0f;                    // re-zeroed on every launch (graph-safe)
    }
}

// Issue one staging step: copy JSTAGE rows x 256 halves of hi and lo into LDS buffer.
// Each of the 256 threads: row r = tid>>2, quarter q = tid&3 -> 128 B per array,
// as 8 async b128 copies (GLOBAL_LOAD_ASYNC_TO_LDS_B128, tracked by ASYNCcnt).
__device__ __forceinline__ void stage_async(const _Float16* __restrict__ xhi,
                                            const _Float16* __restrict__ xlo,
                                            _Float16* ldsHiBuf, _Float16* ldsLoBuf,
                                            int jrow0, int tid)
{
    const int r = tid >> 2;
    const int q = tid & 3;
    const _Float16* shi = xhi + (size_t)(jrow0 + r) * D + q * 64;
    const _Float16* slo = xlo + (size_t)(jrow0 + r) * D + q * 64;
    const unsigned dhi = (unsigned)(uintptr_t)(ldsHiBuf + r * LDSTRIDE + q * 64);
    const unsigned dlo = (unsigned)(uintptr_t)(ldsLoBuf + r * LDSTRIDE + q * 64);
    #pragma unroll
    for (int c = 0; c < 8; ++c) {
        asm volatile("global_load_async_to_lds_b128 %0, %1, off"
                     :: "v"(dhi + c * 16), "v"(shi + c * 8) : "memory");
        asm volatile("global_load_async_to_lds_b128 %0, %1, off"
                     :: "v"(dlo + c * 16), "v"(slo + c * 8) : "memory");
    }
}

// ------------- Kernel 2: fused  sum_j exp(2 * xn_i . xn_j)  via f16-split WMMA -------------
__global__ __launch_bounds__(256) void k_simexp(const _Float16* __restrict__ xhi,
                                                const _Float16* __restrict__ xlo,
                                                float* __restrict__ rowsum)
{
    __shared__ __align__(16) _Float16 ldsHi[2][JSTAGE * LDSTRIDE];   // 2 x 33 KB
    __shared__ __align__(16) _Float16 ldsLo[2][JSTAGE * LDSTRIDE];   // 2 x 33 KB

    const int tid   = threadIdx.x;
    const int lane  = tid & 31;
    const int wave  = tid >> 5;                        // 0..7
    const int i0    = blockIdx.x * ITILE + wave * 16;  // this wave's 16 rows
    const int jbase = blockIdx.y * JCHUNK;

    const int lmod = lane & 15;
    const int lhi  = lane >> 4;                        // 0 or 1 (K-half select)

    // ---- A fragments for all K (rows i0..i0+15), kept in registers ----
    // A layout (16-bit 16x32, ISA 7.12.2): lane L holds row M=L%16,
    //   v16h elems 0..7  = K(k0 + 8*(L/16)     .. +7)
    //   v16h elems 8..15 = K(k0 + 16 + 8*(L/16).. +7)
    v16h Ahi[8], Alo[8];
    {
        const _Float16* ah = xhi + (size_t)(i0 + lmod) * D;
        const _Float16* al = xlo + (size_t)(i0 + lmod) * D;
        #pragma unroll
        for (int ks = 0; ks < 8; ++ks) {
            const int k0 = ks * 32;
            v8h h0 = *(const v8h*)(ah + k0 + lhi * 8);
            v8h h1 = *(const v8h*)(ah + k0 + 16 + lhi * 8);
            Ahi[ks] = __builtin_shufflevector(h0, h1, 0,1,2,3,4,5,6,7,8,9,10,11,12,13,14,15);
            v8h l0 = *(const v8h*)(al + k0 + lhi * 8);
            v8h l1 = *(const v8h*)(al + k0 + 16 + lhi * 8);
            Alo[ks] = __builtin_shufflevector(l0, l1, 0,1,2,3,4,5,6,7,8,9,10,11,12,13,14,15);
        }
    }

    // Per-lane partial sums of exp; elem r corresponds to row i0 + 8*lhi + r.
    v8f sums = {};

    const int G = JCHUNK / JSTAGE;                     // 16 staging steps
    stage_async(xhi, xlo, ldsHi[0], ldsLo[0], jbase, tid);

    for (int g = 0; g < G; ++g) {
        const int buf = g & 1;
        asm volatile("s_wait_asynccnt 0" ::: "memory"); // our stage-g copies landed in LDS
        __syncthreads();                                // everyone's copies landed
        if (g + 1 < G)                                  // overlap next stage with compute
            stage_async(xhi, xlo, ldsHi[buf ^ 1], ldsLo[buf ^ 1],
                        jbase + (g + 1) * JSTAGE, tid);

        #pragma unroll
        for (int jt = 0; jt < JSTAGE / 16; ++jt) {
            // B layout (16-bit 32x16): lane L = column N=L%16, elems 0..7 = K(k0 + 8*(L/16) ..+7), elems 8..15 = K(k0 + 16 + 8*(L/16) ..+7)
            const _Float16* bh = &ldsHi[buf][(jt * 16 + lmod) * LDSTRIDE];
            const _Float16* bl = &ldsLo[buf][(jt * 16 + lmod) * LDSTRIDE];

            v8f acc = {};
            #pragma unroll
            for (int ks = 0; ks < 8; ++ks) {
                const int k0 = ks * 32;
                v8h bh0 = *(const v8h*)(bh + k0 + lhi * 8);
                v8h bh1 = *(const v8h*)(bh + k0 + 16 + lhi * 8);
                v16h Bhi = __builtin_shufflevector(bh0, bh1, 0,1,2,3,4,5,6,7,8,9,10,11,12,13,14,15);
                v8h bl0 = *(const v8h*)(bl + k0 + lhi * 8);
                v8h bl1 = *(const v8h*)(bl + k0 + 16 + lhi * 8);
                v16h Blo = __builtin_shufflevector(bl0, bl1, 0,1,2,3,4,5,6,7,8,9,10,11,12,13,14,15);

                // sim ~= hi*hi + hi*lo + lo*hi   (lo*lo ~ 2^-22, negligible)
                acc = __builtin_amdgcn_wmma_f32_16x16x32_f16(false, Ahi[ks], false, Bhi,
                                                             (short)0, acc, false, false);
                acc = __builtin_amdgcn_wmma_f32_16x16x32_f16(false, Ahi[ks], false, Blo,
                                                             (short)0, acc, false, false);
                acc = __builtin_amdgcn_wmma_f32_16x16x32_f16(false, Alo[ks], false, Bhi,
                                                             (short)0, acc, false, false);
            }

            // exp(sim/T) = exp2(sim * 2*log2(e)); logits in [-2,2] -> no overflow, no max pass.
            #pragma unroll
            for (int r = 0; r < 8; ++r)
                sums[r] += __builtin_exp2f(acc[r] * (TEMP_INV * LOG2E));
        }
    }

    // Reduce over the 16 columns (lanes within each half) -> per-row sums.
    #pragma unroll
    for (int r = 0; r < 8; ++r) {
        float s = sums[r];
        #pragma unroll
        for (int m = 1; m < 16; m <<= 1)
            s += __shfl_xor(s, m, 32);
        sums[r] = s;
    }
    if (lmod == 0) {
        #pragma unroll
        for (int r = 0; r < 8; ++r) {
            const int row = i0 + lhi * 8 + r;
            __hip_atomic_fetch_add(&rowsum[row], sums[r],
                                   __ATOMIC_RELAXED, __HIP_MEMORY_SCOPE_AGENT);
        }
    }
}

// ------------- Kernel 3: loss = mean_i [ log(rowsum_i) - diag_i / T ] -------------
__global__ __launch_bounds__(256) void k_loss(const float* __restrict__ rowsum,
                                              const float* __restrict__ diag,
                                              float* __restrict__ out)
{
    __shared__ float red[8];
    const int t = threadIdx.x;
    float acc = 0.f;
    for (int i = t; i < N; i += 256)
        acc += logf(rowsum[i]) - diag[i] * TEMP_INV;
    #pragma unroll
    for (int m = 16; m >= 1; m >>= 1)
        acc += __shfl_xor(acc, m, 32);
    if ((t & 31) == 0) red[t >> 5] = acc;
    __syncthreads();
    if (t < 8) {
        float v = red[t];
        #pragma unroll
        for (int m = 4; m >= 1; m >>= 1)
            v += __shfl_xor(v, m, 8);
        if (t == 0) out[0] = v / (float)N;
    }
}

extern "C" void kernel_launch(void* const* d_in, const int* in_sizes, int n_in,
                              void* d_out, int out_size, void* d_ws, size_t ws_size,
                              hipStream_t stream)
{
    const float* x = (const float*)d_in[0];
    float* out = (float*)d_out;

    char* ws = (char*)d_ws;
    _Float16* xhi   = (_Float16*)ws;                                   // 4 MB
    _Float16* xlo   = (_Float16*)(ws + (size_t)N * D * 2);             // 4 MB
    float*    rowsum = (float*)(ws + (size_t)N * D * 4);               // 32 KB
    float*    diag   = (float*)(ws + (size_t)N * D * 4 + (size_t)N * 4); // 32 KB

    k_normalize<<<N, 32, 0, stream>>>(x, xhi, xlo, diag, rowsum);

    dim3 grid(N / ITILE, N / JCHUNK);   // 64 x 8 = 512 blocks, 256 thr (8 waves)
    k_simexp<<<grid, 256, 0, stream>>>(xhi, xlo, rowsum);

    k_loss<<<1, 256, 0, stream>>>(rowsum, diag, out);
}
